// SizeInvariantTimeSformer_91087666414101
// MI455X (gfx1250) — compile-verified, matrix operand order checked by reading, not run on hardware
//
#include <hip/hip_runtime.h>
#include <math.h>

// ---------------------------------------------------------------------------
// SizeInvariantTimeSformer forward for MI455X (gfx1250, wave32, WMMA).
// Heavy GEMMs: v_wmma_f32_16x16x32_bf16 with GLOBAL_LOAD_ASYNC_TO_LDS_B128
// double-buffered staging (ASYNCcnt pipeline). Space attention is a
// register-resident flash-attention built on the same WMMA op.
// ---------------------------------------------------------------------------

#define B_      4
#define F_      16
#define CH_     1280
#define NPATCH_ 196
#define DIM_    1024
#define HEADS_  16
#define DH_     64
#define SEQ_    3137
#define MROWS_  (B_ * SEQ_)          // 12548
#define TOKROWS_ (B_ * F_ * NPATCH_) // 12544
#define SCALE_  0.125f               // 64^-0.5

typedef __bf16 bf16_t;
typedef bf16_t v16bf __attribute__((ext_vector_type(16)));
typedef float  v8f   __attribute__((ext_vector_type(8)));

union FragAB { v16bf v; uint4 q[2]; };
union FragC  { v8f   v; float f[8]; };

__device__ __forceinline__ unsigned short f2bf(float f) {
    unsigned int u = __float_as_uint(f);
    unsigned int r = u + 0x7FFFu + ((u >> 16) & 1u);   // round-to-nearest-even
    return (unsigned short)(r >> 16);
}
__device__ __forceinline__ float bf2f(unsigned short h) {
    return __uint_as_float(((unsigned int)h) << 16);
}

// per-lane async DMA of 16B global -> LDS (tracked by ASYNCcnt)
#define ASYNC_B128(ldsoff, gptr)                                             \
    asm volatile("global_load_async_to_lds_b128 %0, %1, off"                 \
                 :: "v"(ldsoff), "v"(gptr) : "memory")
#define WAIT_ASYNC0() asm volatile("s_wait_asynccnt 0" ::: "memory")

// ---------------------------------------------------------------------------
// Tiled bf16 GEMM: out[M,N] = A[M,K] @ Wt[N,K]^T (+bias)(+resid)
// BM=BN=128, BK=32. 256 threads = 8 waves in a 4x2 grid; each wave owns a
// 32x64 C patch = 2x4 wmma tiles. LDS row stride 40 bf16 (20 banks) makes
// the per-lane ds_load_b128 fragment reads bank-conflict free.
// Tile staging uses GLOBAL_LOAD_ASYNC_TO_LDS_B128 double buffering: the DMA
// fills buffer (kt+1) while the wave issues WMMAs on buffer kt; handoff is
// s_wait_asynccnt 0 + s_barrier. OOB A rows are clamped (their C rows are
// discarded by the guarded store).
// row_map==1 remaps output row m -> m + m/3136 + 1 (patch tokens into xt).
// ---------------------------------------------------------------------------
__global__ __launch_bounds__(256)
void gemm_bf16_wmma(const unsigned short* __restrict__ A,
                    const unsigned short* __restrict__ Wt,
                    const float* __restrict__ bias,
                    const float* __restrict__ resid,
                    float* __restrict__ outF,
                    unsigned short* __restrict__ outH,
                    int M, int N, int K, int row_map)
{
    __shared__ unsigned short lA[2][128 * 40];
    __shared__ unsigned short lB[2][128 * 40];

    const int tid  = threadIdx.x;
    const int lane = tid & 31;
    const int wid  = tid >> 5;
    const int wm   = wid & 3;           // 4 waves along M
    const int wn   = wid >> 2;          // 2 waves along N
    const int bm   = blockIdx.x * 128;
    const int bn   = blockIdx.y * 128;
    const int nk   = K >> 5;
    const int half = (lane >> 4) << 3;  // 0 or 8 (A/B fragment K-offset)

    // loader mapping: 1024 x 16B chunks per tile pair, 4 async DMAs per thread
    const int r0 = tid >> 2;
    const int r1 = r0 + 64;
    const int c0 = (tid & 3) << 3;

    const int ga0 = min(bm + r0, M - 1);   // clamp OOB rows (stores guarded)
    const int ga1 = min(bm + r1, M - 1);
    const unsigned short* pa0 = A  + (size_t)ga0 * K + c0;
    const unsigned short* pa1 = A  + (size_t)ga1 * K + c0;
    const unsigned short* pb0 = Wt + (size_t)(bn + r0) * K + c0;
    const unsigned short* pb1 = Wt + (size_t)(bn + r1) * K + c0;

    // low 32 bits of a flat LDS pointer are the LDS byte offset
    const unsigned ldsA0 = (unsigned)(size_t)&lA[0][r0 * 40 + c0];
    const unsigned ldsA1 = (unsigned)(size_t)&lA[0][r1 * 40 + c0];
    const unsigned ldsB0 = (unsigned)(size_t)&lB[0][r0 * 40 + c0];
    const unsigned ldsB1 = (unsigned)(size_t)&lB[0][r1 * 40 + c0];
    const unsigned bufStride = 128u * 40u * 2u;   // bytes between LDS buffers

    FragC acc[2][4];
#pragma unroll
    for (int i = 0; i < 2; ++i)
#pragma unroll
        for (int j = 0; j < 4; ++j)
#pragma unroll
            for (int r = 0; r < 8; ++r) acc[i][j].f[r] = 0.f;

    // prologue: DMA tile 0 into buffer 0
    ASYNC_B128(ldsA0, pa0);
    ASYNC_B128(ldsA1, pa1);
    ASYNC_B128(ldsB0, pb0);
    ASYNC_B128(ldsB1, pb1);
    WAIT_ASYNC0();
    __syncthreads();

    for (int kt = 0; kt < nk; ++kt) {
        const int buf = kt & 1;
        if (kt + 1 < nk) {   // DMA next K-tile into the other buffer
            const unsigned boff = (unsigned)(buf ^ 1) * bufStride;
            const size_t ko = (size_t)(kt + 1) << 5;
            ASYNC_B128(ldsA0 + boff, pa0 + ko);
            ASYNC_B128(ldsA1 + boff, pa1 + ko);
            ASYNC_B128(ldsB0 + boff, pb0 + ko);
            ASYNC_B128(ldsB1 + boff, pb1 + ko);
        }
        if (kt + 2 < nk)     // warm L2 for the weight stream (global_prefetch)
            __builtin_prefetch(pb0 + ((size_t)(kt + 2) << 5), 0, 3);

        FragAB a[2], b[4];
#pragma unroll
        for (int i = 0; i < 2; ++i) {
            const unsigned short* p = &lA[buf][(wm * 32 + i * 16 + (lane & 15)) * 40 + half];
            a[i].q[0] = *(const uint4*)p;
            a[i].q[1] = *(const uint4*)(p + 16);
        }
#pragma unroll
        for (int j = 0; j < 4; ++j) {
            const unsigned short* p = &lB[buf][(wn * 64 + j * 16 + (lane & 15)) * 40 + half];
            b[j].q[0] = *(const uint4*)p;
            b[j].q[1] = *(const uint4*)(p + 16);
        }
#pragma unroll
        for (int i = 0; i < 2; ++i)
#pragma unroll
            for (int j = 0; j < 4; ++j)
                acc[i][j].v = __builtin_amdgcn_wmma_f32_16x16x32_bf16(
                    false, a[i].v, false, b[j].v, (short)0, acc[i][j].v, false, false);

        WAIT_ASYNC0();       // own DMAs into buf^1 have landed
        __syncthreads();     // whole workgroup's DMAs visible; buf reads done
    }

    // Epilogue. C layout: VGPR r holds row r (lanes 0-15) / row r+8 (lanes 16-31).
#pragma unroll
    for (int i = 0; i < 2; ++i) {
#pragma unroll
        for (int j = 0; j < 4; ++j) {
#pragma unroll
            for (int r = 0; r < 8; ++r) {
                const int row = bm + wm * 32 + i * 16 + r + ((lane >> 4) << 3);
                if (row < M) {
                    const int col = bn + wn * 64 + j * 16 + (lane & 15);
                    float v = acc[i][j].f[r];
                    if (bias) v += bias[col];
                    int orow = row;
                    if (row_map) orow = row + row / 3136 + 1;
                    if (resid) v += resid[(size_t)orow * N + col];
                    if (outF) outF[(size_t)orow * N + col] = v;
                    else      outH[(size_t)orow * N + col] = f2bf(v);
                }
            }
        }
    }
}

// ---------------------------------------------------------------------------
// Space attention per (b,h,f): 196 queries x 197 keys (cls prefix) x 64 dims.
// Flash-style: Q fragments straight from global, K fragments from global
// (L2-resident), V^T staged in LDS; online softmax in registers with
// 16-lane __shfl_xor reductions; P relaid out through a tiny per-wave LDS
// tile to become the A fragment for the PV wmma.
// ---------------------------------------------------------------------------
#define VSTR_ 232   // V^T row stride in bf16 (116-bank stride -> conflict-free)

__global__ __launch_bounds__(256)
void space_attn_kernel(const unsigned short* __restrict__ qkv,
                       unsigned short* __restrict__ attnout)
{
    __shared__ unsigned short Vt[64 * VSTR_];     // ~29.7 KB
    __shared__ unsigned short Pt[8][16 * 40];     // 1.25 KB per wave

    const int blk  = blockIdx.x;                  // (b*16 + h)*16 + f
    const int f    = blk & 15;
    const int h    = (blk >> 4) & 15;
    const int b    = blk >> 8;
    const int tid  = threadIdx.x;
    const int lane = tid & 31;
    const int wid  = tid >> 5;
    const int half = (lane >> 4) << 3;
    const size_t base = (size_t)b * SEQ_ * 3072;
    const size_t tok0 = (size_t)(1 + f * NPATCH_);

    // stage V^T[dim][key] (keys padded to 224 with zeros)
    for (int i = tid; i < 64 * 224; i += 256) {
        const int key = i % 224, d = i / 224;
        unsigned short v = 0;
        if (key < 197) {
            const size_t tok = (key == 0) ? 0 : tok0 + key - 1;
            v = qkv[base + tok * 3072 + 2048 + h * 64 + d];
        }
        Vt[d * VSTR_ + key] = v;
    }
    __syncthreads();

    for (int qb = wid; qb < 13; qb += 8) {        // 13 blocks of 16 queries
        int qr = qb * 16 + (lane & 15);
        if (qr > 195) qr = 195;                   // clamp (stores guarded)
        const unsigned short* qp = qkv + base + (tok0 + qr) * 3072 + h * 64;
        FragAB qa[2];
#pragma unroll
        for (int kk = 0; kk < 2; ++kk) {
            qa[kk].q[0] = *(const uint4*)(qp + kk * 32 + half);
            qa[kk].q[1] = *(const uint4*)(qp + kk * 32 + half + 16);
        }

        float m[8], l[8];
        FragC o[4];
#pragma unroll
        for (int r = 0; r < 8; ++r) { m[r] = -3.4e38f; l[r] = 0.f; }
#pragma unroll
        for (int nt = 0; nt < 4; ++nt)
#pragma unroll
            for (int r = 0; r < 8; ++r) o[nt].f[r] = 0.f;

        for (int dj = 0; dj < 7; ++dj) {          // 32 keys per iteration
            FragC c[2];
#pragma unroll
            for (int g = 0; g < 2; ++g) {
#pragma unroll
                for (int r = 0; r < 8; ++r) c[g].f[r] = 0.f;
                const int keyrow = dj * 32 + g * 16 + (lane & 15);
                size_t ktok = 0;
                if (keyrow >= 1 && keyrow < 197) ktok = tok0 + keyrow - 1;
                const unsigned short* kp = qkv + base + ktok * 3072 + 1024 + h * 64;
#pragma unroll
                for (int kk = 0; kk < 2; ++kk) {
                    FragAB kb;
                    kb.q[0] = *(const uint4*)(kp + kk * 32 + half);
                    kb.q[1] = *(const uint4*)(kp + kk * 32 + half + 16);
                    c[g].v = __builtin_amdgcn_wmma_f32_16x16x32_bf16(
                        false, qa[kk].v, false, kb.v, (short)0, c[g].v, false, false);
                }
            }
            // online softmax update over the 32 new key columns
#pragma unroll
            for (int r = 0; r < 8; ++r) {
                const int k0 = dj * 32 + (lane & 15);
                float s0 = (k0 < 197)      ? c[0].f[r] * SCALE_ : -3.4e38f;
                float s1 = (k0 + 16 < 197) ? c[1].f[r] * SCALE_ : -3.4e38f;
                float t = fmaxf(s0, s1);
#pragma unroll
                for (int xm = 1; xm < 16; xm <<= 1) t = fmaxf(t, __shfl_xor(t, xm, 32));
                const float nm = fmaxf(m[r], t);
                const float p0 = __expf(s0 - nm);
                const float p1 = __expf(s1 - nm);
                float ps = p0 + p1;
#pragma unroll
                for (int xm = 1; xm < 16; xm <<= 1) ps += __shfl_xor(ps, xm, 32);
                const float corr = __expf(m[r] - nm);
                l[r] = l[r] * corr + ps;
                m[r] = nm;
#pragma unroll
                for (int nt = 0; nt < 4; ++nt) o[nt].f[r] *= corr;
                const int prow = r + ((lane >> 4) << 3);
                Pt[wid][prow * 40 + (lane & 15)]      = f2bf(p0);
                Pt[wid][prow * 40 + 16 + (lane & 15)] = f2bf(p1);
            }
            asm volatile("s_wait_dscnt 0" ::: "memory");   // P stores -> A-frag loads
            FragAB pa;
            {
                const unsigned short* pp = &Pt[wid][(lane & 15) * 40 + half];
                pa.q[0] = *(const uint4*)pp;
                pa.q[1] = *(const uint4*)(pp + 16);
            }
#pragma unroll
            for (int nt = 0; nt < 4; ++nt) {
                FragAB vb;
                const unsigned short* vp = &Vt[(nt * 16 + (lane & 15)) * VSTR_ + dj * 32 + half];
                vb.q[0] = *(const uint4*)vp;
                vb.q[1] = *(const uint4*)(vp + 16);
                o[nt].v = __builtin_amdgcn_wmma_f32_16x16x32_bf16(
                    false, pa.v, false, vb.v, (short)0, o[nt].v, false, false);
            }
            asm volatile("s_wait_dscnt 0" ::: "memory");   // reads done before next overwrite
        }
#pragma unroll
        for (int nt = 0; nt < 4; ++nt) {
#pragma unroll
            for (int r = 0; r < 8; ++r) {
                const int qrow = qb * 16 + r + ((lane >> 4) << 3);
                if (qrow < 196) {
                    const float v = o[nt].f[r] / l[r];
                    attnout[((size_t)b * SEQ_ + 1 + f * NPATCH_ + qrow) * 1024 +
                            h * 64 + nt * 16 + (lane & 15)] = f2bf(v);
                }
            }
        }
    }
}

// ---------------------------------------------------------------------------
// Time attention: one thread per query (b,h,n,f): 16 frames x 17 keys x 64.
// ---------------------------------------------------------------------------
__global__ __launch_bounds__(256)
void time_attn_kernel(const unsigned short* __restrict__ qkv,
                      const unsigned char* __restrict__ mask,
                      const unsigned char* __restrict__ idm,
                      unsigned short* __restrict__ attnout)
{
    const int idx = blockIdx.x * 256 + threadIdx.x;
    if (idx >= B_ * HEADS_ * NPATCH_ * F_) return;
    const int f = idx & 15;
    const int n = (idx >> 4) % NPATCH_;
    const int h = (idx / (16 * NPATCH_)) & 15;
    const int b = idx / (16 * NPATCH_ * 16);
    const size_t base = (size_t)b * SEQ_ * 3072;
    const size_t qtok = 1 + (size_t)f * NPATCH_ + n;

    float q[64];
#pragma unroll
    for (int d = 0; d < 64; ++d) q[d] = bf2f(qkv[base + qtok * 3072 + h * 64 + d]);

    float sim[17];
    size_t voff[17];
#pragma unroll 1
    for (int j = 0; j < 17; ++j) {
        const size_t tok = (j == 0) ? 0 : 1 + (size_t)(j - 1) * NPATCH_ + n;
        const unsigned short* kp = qkv + base + tok * 3072 + 1024 + h * 64;
        voff[j] = base + tok * 3072 + 2048 + h * 64;
        float s = 0.f;
#pragma unroll
        for (int d = 0; d < 64; ++d) s += q[d] * bf2f(kp[d]);
        const bool ok = (j == 0) ||
                        (mask[b * 16 + (j - 1)] && idm[(b * 16 + f) * 16 + (j - 1)]);
        sim[j] = ok ? s * SCALE_ : -3.4e38f;
    }
    float mx = sim[0];
#pragma unroll
    for (int j = 1; j < 17; ++j) mx = fmaxf(mx, sim[j]);
    float sum = 0.f;
#pragma unroll
    for (int j = 0; j < 17; ++j) { sim[j] = __expf(sim[j] - mx); sum += sim[j]; }
    const float inv = 1.f / sum;
#pragma unroll 1
    for (int d = 0; d < 64; ++d) {
        float o = 0.f;
#pragma unroll
        for (int j = 0; j < 17; ++j) o += sim[j] * bf2f(qkv[voff[j] + d]);
        attnout[((size_t)b * SEQ_ + qtok) * 1024 + h * 64 + d] = f2bf(o * inv);
    }
}

// ---------------------------------------------------------------------------
// Cls-token attention per (b,h): 1 query over all 3137 tokens.
// ---------------------------------------------------------------------------
__global__ __launch_bounds__(256)
void cls_attn_kernel(const unsigned short* __restrict__ qkv,
                     const unsigned char* __restrict__ mask,
                     unsigned short* __restrict__ attnout)
{
    __shared__ float qsh[64];
    __shared__ float sims[SEQ_];
    __shared__ float red[256];
    __shared__ float shm, shl;
    const int tid = threadIdx.x;
    const int b = blockIdx.x >> 4, h = blockIdx.x & 15;
    const size_t base = (size_t)b * SEQ_ * 3072;
    if (tid < 64) qsh[tid] = bf2f(qkv[base + h * 64 + tid]);
    __syncthreads();

    float lmax = -3.4e38f;
    for (int s2 = tid; s2 < SEQ_; s2 += 256) {
        const unsigned short* kp = qkv + base + (size_t)s2 * 3072 + 1024 + h * 64;
        float s = 0.f;
#pragma unroll
        for (int d = 0; d < 64; ++d) s += qsh[d] * bf2f(kp[d]);
        const bool ok = (s2 == 0) || mask[b * 16 + (s2 - 1) / NPATCH_];
        s = ok ? s * SCALE_ : -3.4e38f;
        sims[s2] = s;
        lmax = fmaxf(lmax, s);
    }
    red[tid] = lmax; __syncthreads();
    for (int st = 128; st > 0; st >>= 1) {
        if (tid < st) red[tid] = fmaxf(red[tid], red[tid + st]);
        __syncthreads();
    }
    if (tid == 0) shm = red[0];
    __syncthreads();
    float lsum = 0.f;
    for (int s2 = tid; s2 < SEQ_; s2 += 256) {
        const float p = __expf(sims[s2] - shm);
        sims[s2] = p;
        lsum += p;
    }
    red[tid] = lsum; __syncthreads();
    for (int st = 128; st > 0; st >>= 1) {
        if (tid < st) red[tid] += red[tid + st];
        __syncthreads();
    }
    if (tid == 0) shl = red[0];
    __syncthreads();
    const int d = tid & 63, part = tid >> 6;
    float acc = 0.f;
    for (int s2 = part; s2 < SEQ_; s2 += 4)
        acc += sims[s2] * bf2f(qkv[base + (size_t)s2 * 3072 + 2048 + h * 64 + d]);
    red[tid] = acc; __syncthreads();
    if (tid < 64) {
        const float o = red[tid] + red[tid + 64] + red[tid + 128] + red[tid + 192];
        attnout[((size_t)b * SEQ_) * 1024 + h * 64 + tid] = f2bf(o / shl);
    }
}

// ---------------------------------------------------------------------------
// LayerNorm: one wave32 per 1024-wide row, bf16 output.
// ---------------------------------------------------------------------------
__global__ __launch_bounds__(256)
void layernorm_bf16(const float* __restrict__ x, const float* __restrict__ g,
                    const float* __restrict__ bta, unsigned short* __restrict__ out,
                    int rows)
{
    const int row = blockIdx.x * 8 + (threadIdx.x >> 5);
    const int lane = threadIdx.x & 31;
    if (row >= rows) return;
    const float* xr = x + (size_t)row * 1024;
    float s = 0.f;
    for (int i = lane; i < 1024; i += 32) s += xr[i];
#pragma unroll
    for (int xm = 1; xm < 32; xm <<= 1) s += __shfl_xor(s, xm, 32);
    const float mean = s * (1.f / 1024.f);
    float v = 0.f;
    for (int i = lane; i < 1024; i += 32) { const float d = xr[i] - mean; v += d * d; }
#pragma unroll
    for (int xm = 1; xm < 32; xm <<= 1) v += __shfl_xor(v, xm, 32);
    const float rstd = rsqrtf(v * (1.f / 1024.f) + 1e-5f);
    for (int i = lane; i < 1024; i += 32) {
        const float y = (xr[i] - mean) * rstd * g[i] + bta[i];
        out[(size_t)row * 1024 + i] = f2bf(y);
    }
}

// GLU epilogue: act = a * gelu_exact(g)
__global__ void glu_kernel(const unsigned short* __restrict__ hh,
                           unsigned short* __restrict__ act)
{
    const size_t idx = (size_t)blockIdx.x * 256 + threadIdx.x;
    if (idx >= (size_t)MROWS_ * 4096) return;
    const int j = (int)(idx & 4095);
    const size_t mrow = idx >> 12;
    const float a = bf2f(hh[mrow * 8192 + j]);
    const float g = bf2f(hh[mrow * 8192 + 4096 + j]);
    const float ge = 0.5f * g * (1.f + erff(g * 0.70710678118654752f));
    act[idx] = f2bf(a * ge);
}

// x [B,F,CH,14,14] -> A_patch bf16 [12544, 1280] (channel-last rows)
__global__ void im2row_kernel(const float* __restrict__ x,
                              unsigned short* __restrict__ dst)
{
    const size_t idx = (size_t)blockIdx.x * 256 + threadIdx.x;
    if (idx >= (size_t)TOKROWS_ * CH_) return;
    const int c = (int)(idx % CH_);
    const int mrow = (int)(idx / CH_);
    const int b = mrow / 3136, r = mrow % 3136;
    const int f = r / NPATCH_, p = r % NPATCH_;
    dst[idx] = f2bf(x[((size_t)(b * 16 + f) * CH_ + c) * NPATCH_ + p]);
}

// set cls row, add positional embedding
__global__ void pos_cls_kernel(float* __restrict__ xt, const float* __restrict__ cls,
                               const float* __restrict__ pos)
{
    const size_t idx = (size_t)blockIdx.x * 256 + threadIdx.x;
    if (idx >= (size_t)MROWS_ * 1024) return;
    const int d = (int)(idx & 1023);
    const int srow = (int)(idx >> 10);
    const int b = srow / SEQ_;
    const int r = srow - b * SEQ_;
    if (r == 0) xt[idx] = cls[d] + pos[d];
    else        xt[idx] += pos[(size_t)r * 1024 + d];
}

// W[K,N] fp32 -> Wt[N,K] bf16
__global__ void transpose_to_bf16(const float* __restrict__ src,
                                  unsigned short* __restrict__ dst, int K, int N)
{
    const size_t idx = (size_t)blockIdx.x * 256 + threadIdx.x;
    if (idx >= (size_t)K * N) return;
    const int n = (int)(idx / K);
    const int k = (int)(idx % K);
    dst[idx] = f2bf(src[(size_t)k * N + n]);
}

// final head: out[b] = LN(xt[b,0]) . o_w + o_b
__global__ __launch_bounds__(256)
void head_kernel(const float* __restrict__ xt, const float* __restrict__ g,
                 const float* __restrict__ bta, const float* __restrict__ ow,
                 const float* __restrict__ ob, float* __restrict__ out)
{
    __shared__ float red[256];
    __shared__ float s_mean, s_rstd;
    const int tid = threadIdx.x, b = blockIdx.x;
    const float* xr = xt + (size_t)b * SEQ_ * 1024;
    float s = 0.f;
    for (int i = tid; i < 1024; i += 256) s += xr[i];
    red[tid] = s; __syncthreads();
    for (int st = 128; st > 0; st >>= 1) { if (tid < st) red[tid] += red[tid + st]; __syncthreads(); }
    if (tid == 0) s_mean = red[0] * (1.f / 1024.f);
    __syncthreads();
    float v = 0.f;
    for (int i = tid; i < 1024; i += 256) { const float d = xr[i] - s_mean; v += d * d; }
    red[tid] = v; __syncthreads();
    for (int st = 128; st > 0; st >>= 1) { if (tid < st) red[tid] += red[tid + st]; __syncthreads(); }
    if (tid == 0) s_rstd = rsqrtf(red[0] * (1.f / 1024.f) + 1e-5f);
    __syncthreads();
    float dacc = 0.f;
    for (int i = tid; i < 1024; i += 256)
        dacc += ((xr[i] - s_mean) * s_rstd * g[i] + bta[i]) * ow[i];
    red[tid] = dacc; __syncthreads();
    for (int st = 128; st > 0; st >>= 1) { if (tid < st) red[tid] += red[tid + st]; __syncthreads(); }
    if (tid == 0) out[b] = red[0] + ob[0];
}

// ---------------------------------------------------------------------------
// Host orchestration
// ---------------------------------------------------------------------------
extern "C" void kernel_launch(void* const* d_in, const int* in_sizes, int n_in,
                              void* d_out, int out_size, void* d_ws, size_t ws_size,
                              hipStream_t stream)
{
    (void)in_sizes; (void)n_in; (void)out_size; (void)ws_size;

    const float* x        = (const float*)d_in[0];
    const unsigned char* mask = (const unsigned char*)d_in[1];  // bool arrays, 1B each
    const unsigned char* idm  = (const unsigned char*)d_in[2];
    const float* patch_w  = (const float*)d_in[3];
    const float* patch_b  = (const float*)d_in[4];
    const float* cls_tok  = (const float*)d_in[5];
    const float* pos_emb  = (const float*)d_in[6];
    const float* t_ng = (const float*)d_in[7];
    const float* t_nb = (const float*)d_in[8];
    const float* t_qkv = (const float*)d_in[9];
    const float* t_ow  = (const float*)d_in[10];
    const float* t_ob  = (const float*)d_in[11];
    const float* s_ng = (const float*)d_in[12];
    const float* s_nb = (const float*)d_in[13];
    const float* s_qkv = (const float*)d_in[14];
    const float* s_ow  = (const float*)d_in[15];
    const float* s_ob  = (const float*)d_in[16];
    const float* f_ng = (const float*)d_in[17];
    const float* f_nb = (const float*)d_in[18];
    const float* f_w1 = (const float*)d_in[19];
    const float* f_b1 = (const float*)d_in[20];
    const float* f_w2 = (const float*)d_in[21];
    const float* f_b2 = (const float*)d_in[22];
    const float* o_ng = (const float*)d_in[23];
    const float* o_nb = (const float*)d_in[24];
    const float* o_w  = (const float*)d_in[25];
    const float* o_b  = (const float*)d_in[26];

    char* ws = (char*)d_ws;
    size_t off = 0;
    auto ALLOC = [&](size_t bytes) -> char* {
        char* p = ws + off;
        off = (off + bytes + 255) & ~(size_t)255;
        return p;
    };
    // bf16 weight pool (transposed [N,K])
    unsigned short* w_patch = (unsigned short*)ALLOC((size_t)1024 * 1280 * 2);
    unsigned short* w_tqkv  = (unsigned short*)ALLOC((size_t)2 * 3072 * 1024 * 2);
    unsigned short* w_tow   = (unsigned short*)ALLOC((size_t)2 * 1024 * 1024 * 2);
    unsigned short* w_sqkv  = (unsigned short*)ALLOC((size_t)2 * 3072 * 1024 * 2);
    unsigned short* w_sow   = (unsigned short*)ALLOC((size_t)2 * 1024 * 1024 * 2);
    unsigned short* w_f1    = (unsigned short*)ALLOC((size_t)2 * 8192 * 1024 * 2);
    unsigned short* w_f2    = (unsigned short*)ALLOC((size_t)2 * 1024 * 4096 * 2);
    // activations
    unsigned short* a_patch = (unsigned short*)ALLOC((size_t)TOKROWS_ * CH_ * 2);
    float*          xt      = (float*)         ALLOC((size_t)MROWS_ * 1024 * 4);
    unsigned short* xn      = (unsigned short*)ALLOC((size_t)MROWS_ * 1024 * 2);
    unsigned short* qkvb    = (unsigned short*)ALLOC((size_t)MROWS_ * 3072 * 2);
    unsigned short* attn    = (unsigned short*)ALLOC((size_t)MROWS_ * 1024 * 2);
    unsigned short* hh      = (unsigned short*)ALLOC((size_t)MROWS_ * 8192 * 2);
    unsigned short* act     = (unsigned short*)ALLOC((size_t)MROWS_ * 4096 * 2);

    auto G = [](size_t n) { return (unsigned)((n + 255) / 256); };

    // --- weight prep (bf16 transpose) ---
    transpose_to_bf16<<<G((size_t)1024 * 1280), 256, 0, stream>>>(patch_w, w_patch, 1280, 1024);
    for (int i = 0; i < 2; ++i) {
        transpose_to_bf16<<<G((size_t)3072 * 1024), 256, 0, stream>>>(
            t_qkv + (size_t)i * 1024 * 3072, w_tqkv + (size_t)i * 3072 * 1024, 1024, 3072);
        transpose_to_bf16<<<G((size_t)1024 * 1024), 256, 0, stream>>>(
            t_ow + (size_t)i * 1024 * 1024, w_tow + (size_t)i * 1024 * 1024, 1024, 1024);
        transpose_to_bf16<<<G((size_t)3072 * 1024), 256, 0, stream>>>(
            s_qkv + (size_t)i * 1024 * 3072, w_sqkv + (size_t)i * 3072 * 1024, 1024, 3072);
        transpose_to_bf16<<<G((size_t)1024 * 1024), 256, 0, stream>>>(
            s_ow + (size_t)i * 1024 * 1024, w_sow + (size_t)i * 1024 * 1024, 1024, 1024);
        transpose_to_bf16<<<G((size_t)8192 * 1024), 256, 0, stream>>>(
            f_w1 + (size_t)i * 1024 * 8192, w_f1 + (size_t)i * 8192 * 1024, 1024, 8192);
        transpose_to_bf16<<<G((size_t)1024 * 4096), 256, 0, stream>>>(
            f_w2 + (size_t)i * 4096 * 1024, w_f2 + (size_t)i * 1024 * 4096, 4096, 1024);
    }

    // --- patch embedding ---
    im2row_kernel<<<G((size_t)TOKROWS_ * CH_), 256, 0, stream>>>(x, a_patch);
    gemm_bf16_wmma<<<dim3(98, 8), 256, 0, stream>>>(
        a_patch, w_patch, patch_b, nullptr, xt, nullptr, TOKROWS_, 1024, 1280, 1);
    pos_cls_kernel<<<G((size_t)MROWS_ * 1024), 256, 0, stream>>>(xt, cls_tok, pos_emb);

    const unsigned gm = (MROWS_ + 127) / 128;   // 99
    const unsigned ln_grid = (MROWS_ + 7) / 8;

    for (int i = 0; i < 2; ++i) {
        // ---- time attention block ----
        layernorm_bf16<<<ln_grid, 256, 0, stream>>>(xt, t_ng + i * 1024, t_nb + i * 1024, xn, MROWS_);
        gemm_bf16_wmma<<<dim3(gm, 24), 256, 0, stream>>>(
            xn, w_tqkv + (size_t)i * 3072 * 1024, nullptr, nullptr, nullptr, qkvb,
            MROWS_, 3072, 1024, 0);
        time_attn_kernel<<<784, 256, 0, stream>>>(qkvb, mask, idm, attn);
        cls_attn_kernel<<<64, 256, 0, stream>>>(qkvb, mask, attn);
        gemm_bf16_wmma<<<dim3(gm, 8), 256, 0, stream>>>(
            attn, w_tow + (size_t)i * 1024 * 1024, t_ob + i * 1024, xt, xt, nullptr,
            MROWS_, 1024, 1024, 0);
        // ---- space attention block ----
        layernorm_bf16<<<ln_grid, 256, 0, stream>>>(xt, s_ng + i * 1024, s_nb + i * 1024, xn, MROWS_);
        gemm_bf16_wmma<<<dim3(gm, 24), 256, 0, stream>>>(
            xn, w_sqkv + (size_t)i * 3072 * 1024, nullptr, nullptr, nullptr, qkvb,
            MROWS_, 3072, 1024, 0);
        space_attn_kernel<<<1024, 256, 0, stream>>>(qkvb, attn);
        cls_attn_kernel<<<64, 256, 0, stream>>>(qkvb, mask, attn);
        gemm_bf16_wmma<<<dim3(gm, 8), 256, 0, stream>>>(
            attn, w_sow + (size_t)i * 1024 * 1024, s_ob + i * 1024, xt, xt, nullptr,
            MROWS_, 1024, 1024, 0);
        // ---- GLU feed-forward ----
        layernorm_bf16<<<ln_grid, 256, 0, stream>>>(xt, f_ng + i * 1024, f_nb + i * 1024, xn, MROWS_);
        gemm_bf16_wmma<<<dim3(gm, 64), 256, 0, stream>>>(
            xn, w_f1 + (size_t)i * 8192 * 1024, f_b1 + (size_t)i * 8192, nullptr, nullptr, hh,
            MROWS_, 8192, 1024, 0);
        glu_kernel<<<G((size_t)MROWS_ * 4096), 256, 0, stream>>>(hh, act);
        gemm_bf16_wmma<<<dim3(gm, 8), 256, 0, stream>>>(
            act, w_f2 + (size_t)i * 1024 * 4096, f_b2 + i * 1024, xt, xt, nullptr,
            MROWS_, 1024, 4096, 0);
    }

    head_kernel<<<B_, 256, 0, stream>>>(xt, o_ng, o_nb, o_w, o_b, (float*)d_out);
}
